// SwinUnifiedAttention_455266534008
// MI455X (gfx1250) — matrix operand order repaired and checked by reading, not compiled, not run on hardware
//
#include <hip/hip_runtime.h>
#include <hip/hip_bf16.h>

#define DIM   128
#define HEADS 4
#define HD    32
#define NTOK  49
#define NPAD  64
#define NWIN  64

typedef __attribute__((ext_vector_type(16))) _Float16 v16h;
typedef __attribute__((ext_vector_type(8)))  float    v8f;

// ---- LDS layout (bytes), 160 KB total, phase-aliased -----------------------
#define OFF_X   0            // 64x128 f16  (phase 6+: aliased as O)
#define OFF_W   16384        // 128x128 f16 (Wq/Wk/Wv/Wp sequentially)
#define OFF_Q   49152        // 64x128 f16  (phase 5+: aliased as P low half)
#define OFF_K   65536        // 64x128 f16  (phase 5+: aliased as P high half)
#define OFF_VT  81920        // 128x64 f16  (V transposed: [channel][token])
#define OFF_S   98304        // 4x64x64 f32 scores
#define LDS_BYTES 163840
#define OFF_P   OFF_Q        // 4x64x64 f16 probabilities (overlays Q+K)
#define OFF_O   OFF_X        // 64x128 f16 attention output (overlays X)

// ---- WMMA fragment gathers (CDNA5 16-bit layouts, 16B-aligned LDS reads) ---
// A 16x32: lane m=l&15, hi=l>>4; elems 0..7 -> K = hi*8+0..7, elems 8..15 -> K = 16+hi*8+0..7
__device__ __forceinline__ v16h load_frag_a(const _Float16* base, int pitch,
                                            int row0, int k0, int lane) {
  const int r  = row0 + (lane & 15);
  const int hi = lane >> 4;
  const _Float16* p = base + r * pitch + k0 + hi * 8;
  v16h a;
#pragma unroll
  for (int i = 0; i < 8; ++i) { a[i] = p[i]; a[i + 8] = p[i + 16]; }
  return a;
}

// B 32x16: lane n=l&15, hi=l>>4; elems 0..7 -> K = hi*8+0..7, elems 8..15 -> K = 16+hi*8+0..7
// B column n is a row of the (row-major) source -> pass the source row index.
__device__ __forceinline__ v16h load_frag_b(const _Float16* base, int pitch,
                                            int col0, int k0, int lane) {
  const int c  = col0 + (lane & 15);
  const int hi = lane >> 4;
  const _Float16* p = base + c * pitch + k0 + hi * 8;
  v16h b;
#pragma unroll
  for (int i = 0; i < 8; ++i) { b[i] = p[i]; b[i + 8] = p[i + 16]; }
  return b;
}

__device__ __forceinline__ v8f wmma_f16(v16h a, v16h b, v8f c) {
  return __builtin_amdgcn_wmma_f32_16x16x32_f16(false, a, false, b, (short)0, c,
                                                false, false);
}

// ---- staging loads ---------------------------------------------------------
__device__ __forceinline__ void stage_x(const float* __restrict__ src,
                                        _Float16* dst, int tid) {
  // 49x128 fp32 -> 64x128 f16, zero padded rows
  for (int i = tid; i < NPAD * DIM; i += 256) {
    const int row = i >> 7;
    dst[i] = (row < NTOK) ? (_Float16)src[i] : (_Float16)0.f;
  }
}

__device__ __forceinline__ void stage_w(const float* __restrict__ src,
                                        _Float16* dst, int tid) {
  for (int i = tid; i < DIM * DIM; i += 256) dst[i] = (_Float16)src[i];
}

// ---- X[64x128] @ W^T[128x128] + b, per-wave column tile --------------------
// wave w owns output columns [16w, 16w+16); B frags hoisted & reused over M.
__device__ __forceinline__ void proj_gemm(const _Float16* X, const _Float16* W,
                                          const float* __restrict__ bias,
                                          _Float16* dst, int dst_pitch,
                                          bool transpose, int wave, int lane) {
  const int n0  = wave * 16;
  const int col = lane & 15;
  const int hi  = lane >> 4;
  v16h bf[4];
#pragma unroll
  for (int ks = 0; ks < 4; ++ks) bf[ks] = load_frag_b(W, DIM, n0, ks * 32, lane);
  const float bv = bias[n0 + col];
#pragma unroll
  for (int mt = 0; mt < 4; ++mt) {
    v8f acc = {};
#pragma unroll
    for (int ks = 0; ks < 4; ++ks) {
      v16h a = load_frag_a(X, DIM, mt * 16, ks * 32, lane);
      acc = wmma_f16(a, bf[ks], acc);
    }
#pragma unroll
    for (int r = 0; r < 8; ++r) {
      const float v = acc[r] + bv;
      const int   m = mt * 16 + r + 8 * hi;
      if (!transpose) dst[m * dst_pitch + n0 + col] = (_Float16)v;
      else            dst[(n0 + col) * dst_pitch + m] = (_Float16)v;
    }
  }
}

__global__ __launch_bounds__(256, 1)
void swin_attn_kernel(const float* __restrict__ q_in, const float* __restrict__ k_in,
                      const float* __restrict__ v_in,
                      const float* __restrict__ Wq, const float* __restrict__ bq,
                      const float* __restrict__ Wk, const float* __restrict__ bk,
                      const float* __restrict__ Wv, const float* __restrict__ bv,
                      const float* __restrict__ bias_table,
                      const float* __restrict__ Wp, const float* __restrict__ bp,
                      const float* __restrict__ mask,
                      const int* __restrict__ rel_index,
                      float* __restrict__ out) {
  extern __shared__ __align__(16) char smem[];
  _Float16* Xs  = (_Float16*)(smem + OFF_X);
  _Float16* Wsh = (_Float16*)(smem + OFF_W);
  _Float16* Qs  = (_Float16*)(smem + OFF_Q);
  _Float16* Ks  = (_Float16*)(smem + OFF_K);
  _Float16* VTs = (_Float16*)(smem + OFF_VT);
  float*    Ss  = (float*)   (smem + OFF_S);
  _Float16* Ps  = (_Float16*)(smem + OFF_P);
  _Float16* Os  = (_Float16*)(smem + OFF_O);

  const int tid  = threadIdx.x;
  const int lane = tid & 31;
  const int wave = tid >> 5;
  const int col  = lane & 15;
  const int hi   = lane >> 4;
  const int blk  = blockIdx.x;
  const size_t xbase = (size_t)blk * NTOK * DIM;

  // ---- Phase 1: Q = Xq @ Wq^T + bq ----
  stage_x(q_in + xbase, Xs, tid);
  stage_w(Wq, Wsh, tid);
  __syncthreads();
  proj_gemm(Xs, Wsh, bq, Qs, DIM, false, wave, lane);
  __syncthreads();

  // ---- Phase 2: K = Xk @ Wk^T + bk ----
  stage_x(k_in + xbase, Xs, tid);
  stage_w(Wk, Wsh, tid);
  __syncthreads();
  proj_gemm(Xs, Wsh, bk, Ks, DIM, false, wave, lane);
  __syncthreads();

  // ---- Phase 3: V^T = (Xv @ Wv^T + bv)^T  (stored [channel][token]) ----
  stage_x(v_in + xbase, Xs, tid);
  stage_w(Wv, Wsh, tid);
  __syncthreads();
  proj_gemm(Xs, Wsh, bv, VTs, NPAD, true, wave, lane);
  __syncthreads();

  // ---- Phase 4: S = Q @ K^T per head (f32 scores to LDS) ----
  // wave -> head = wave/2, two N-tiles; B columns are rows of K.
  {
    const int head = wave >> 1;
    const int nt0  = (wave & 1) * 2;
    const int kcol = head * HD;
#pragma unroll
    for (int nt = nt0; nt < nt0 + 2; ++nt) {
      v16h bf = load_frag_b(Ks, DIM, nt * 16, kcol, lane);
#pragma unroll
      for (int mt = 0; mt < 4; ++mt) {
        v16h a   = load_frag_a(Qs, DIM, mt * 16, kcol, lane);
        v8f  acc = {};
        acc = wmma_f16(a, bf, acc);
        float* srow = Ss + (size_t)head * NPAD * NPAD;
#pragma unroll
        for (int r = 0; r < 8; ++r)
          srow[(mt * 16 + r + 8 * hi) * NPAD + nt * 16 + col] = acc[r];
      }
    }
  }
  __syncthreads();

  // ---- Phase 5: softmax(S*scale + rel_bias + shift_mask) -> P (f16) ----
  {
    const float scale = 0.17677669529663687f;  // 1/sqrt(32)
    if (tid < HEADS * NTOK) {
      const int head = tid / NTOK;
      const int row  = tid - head * NTOK;
      float* srow = Ss + ((size_t)head * NPAD + row) * NPAD;
      const int*   ri   = rel_index + row * NTOK;
      const float* mrow = mask + (size_t)(blk % NWIN) * NTOK * NTOK + row * NTOK;
      float mx = -3.0e38f;
      for (int m = 0; m < NTOK; ++m) {
        float s = srow[m] * scale + bias_table[ri[m] * HEADS + head] + mrow[m];
        srow[m] = s;
        mx = fmaxf(mx, s);
      }
      float sum = 0.f;
      for (int m = 0; m < NTOK; ++m) {
        float e = __expf(srow[m] - mx);
        srow[m] = e;
        sum += e;
      }
      const float inv = 1.f / sum;
      _Float16* prow = Ps + ((size_t)head * NPAD + row) * NPAD;
      for (int m = 0; m < NTOK; ++m) prow[m] = (_Float16)(srow[m] * inv);
      for (int m = NTOK; m < NPAD; ++m) prow[m] = (_Float16)0.f;
    } else {
      const int idx = tid - HEADS * NTOK;  // zero padded rows 49..63
      if (idx < HEADS * (NPAD - NTOK)) {
        const int head = idx / (NPAD - NTOK);
        const int row  = NTOK + idx % (NPAD - NTOK);
        _Float16* prow = Ps + ((size_t)head * NPAD + row) * NPAD;
        for (int m = 0; m < NPAD; ++m) prow[m] = (_Float16)0.f;
      }
    }
  }
  __syncthreads();

  // ---- Phase 6: O = P @ V per head (B columns are rows of V^T) ----
  {
    const int head = wave >> 1;
    const int nt   = wave & 1;          // 16-col tile inside the 32-wide head
    const int vrow = head * HD + nt * 16;
    v16h b0 = load_frag_b(VTs, NPAD, vrow, 0,  lane);
    v16h b1 = load_frag_b(VTs, NPAD, vrow, 32, lane);
    const _Float16* Ph = Ps + (size_t)head * NPAD * NPAD;
#pragma unroll
    for (int mt = 0; mt < 4; ++mt) {
      v8f acc = {};
      v16h a0 = load_frag_a(Ph, NPAD, mt * 16, 0,  lane);
      acc = wmma_f16(a0, b0, acc);
      v16h a1 = load_frag_a(Ph, NPAD, mt * 16, 32, lane);
      acc = wmma_f16(a1, b1, acc);
#pragma unroll
      for (int r = 0; r < 8; ++r)
        Os[(mt * 16 + r + 8 * hi) * DIM + vrow + col] = (_Float16)acc[r];
    }
  }
  __syncthreads();

  // ---- Phase 7: out = O @ Wp^T + bp (fp32 to global) ----
  stage_w(Wp, Wsh, tid);
  __syncthreads();
  {
    const int n0 = wave * 16;
    v16h bf[4];
#pragma unroll
    for (int ks = 0; ks < 4; ++ks) bf[ks] = load_frag_b(Wsh, DIM, n0, ks * 32, lane);
    const float bv = bp[n0 + col];
#pragma unroll
    for (int mt = 0; mt < 4; ++mt) {
      v8f acc = {};
#pragma unroll
      for (int ks = 0; ks < 4; ++ks) {
        v16h a = load_frag_a(Os, DIM, mt * 16, ks * 32, lane);
        acc = wmma_f16(a, bf[ks], acc);
      }
#pragma unroll
      for (int r = 0; r < 8; ++r) {
        const int m = mt * 16 + r + 8 * hi;
        if (m < NTOK)
          out[xbase + (size_t)m * DIM + n0 + col] = acc[r] + bv;
      }
    }
  }
}

extern "C" void kernel_launch(void* const* d_in, const int* in_sizes, int n_in,
                              void* d_out, int out_size, void* d_ws, size_t ws_size,
                              hipStream_t stream) {
  const float* q_in = (const float*)d_in[0];
  const float* k_in = (const float*)d_in[1];
  const float* v_in = (const float*)d_in[2];
  const float* Wq   = (const float*)d_in[3];
  const float* bq   = (const float*)d_in[4];
  const float* Wk   = (const float*)d_in[5];
  const float* bk   = (const float*)d_in[6];
  const float* Wv   = (const float*)d_in[7];
  const float* bv   = (const float*)d_in[8];
  const float* bt   = (const float*)d_in[9];
  const float* Wp   = (const float*)d_in[10];
  const float* bp   = (const float*)d_in[11];
  const float* mask = (const float*)d_in[12];
  const int*   ridx = (const int*)d_in[13];
  float* out = (float*)d_out;

  const int B = in_sizes[0] / (NTOK * DIM);  // 4096 windows
  swin_attn_kernel<<<B, 256, LDS_BYTES, stream>>>(
      q_in, k_in, v_in, Wq, bq, Wk, bk, Wv, bv, bt, Wp, bp, mask, ridx, out);
}
